// DFRscore_89060441850523
// MI455X (gfx1250) — matrix operand order repaired and checked by reading, not compiled, not run on hardware
//
#include <hip/hip_runtime.h>
#include <hip/hip_bf16.h>

#define ALPHA   0.2f
#define NEG_INF -1000000000.0f

typedef __attribute__((ext_vector_type(16))) _Float16 v16h;
typedef __attribute__((ext_vector_type(8)))  _Float16 v8h;
typedef __attribute__((ext_vector_type(8)))  float    v8f;

// ---- WMMA fragment loaders (CDNA5 ISA 7.12.2 layouts) ----------------------
// A (16x32 f16, MxK): lanes 0-15 -> K = k0+0..7 (elems 0-7), k0+16..23 (elems 8-15)
//                     lanes16-31 -> K = k0+8..15,            k0+24..31
__device__ __forceinline__ v16h frag_a(const _Float16* p, int stride, int row0,
                                       int k0, int lane) {
    int r  = row0 + (lane & 15);
    int kk = k0 + ((lane & 16) ? 8 : 0);
    const _Float16* q = p + r * stride + kk;
    v8h lo = *(const v8h*)(q);
    v8h hi = *(const v8h*)(q + 16);
    v16h f;
#pragma unroll
    for (int i = 0; i < 8; ++i) { f[i] = lo[i]; f[i + 8] = hi[i]; }
    return f;
}
// B (32x16 f16, KxN): lane n = lane&15; lanes 0-15 hold K = k0+0..7, k0+16..23,
//                     lanes 16-31 hold K = k0+8..15, k0+24..31 (same as A).
__device__ __forceinline__ v16h frag_b(const _Float16* p, int stride, int row0,
                                       int k0, int lane) {
    int r  = row0 + (lane & 15);
    int kk = k0 + ((lane & 16) ? 8 : 0);
    const _Float16* q = p + r * stride + kk;
    v8h lo = *(const v8h*)(q);
    v8h hi = *(const v8h*)(q + 16);
    v16h f;
#pragma unroll
    for (int i = 0; i < 8; ++i) { f[i] = lo[i]; f[i + 8] = hi[i]; }
    return f;
}

__device__ __forceinline__ v8f wmma_f32(v16h a, v16h b, v8f c) {
    return __builtin_amdgcn_wmma_f32_16x16x32_f16(false, a, false, b,
                                                  (short)0, c, false, false);
}

// ---- one-shot weight conversion to f16 -------------------------------------
// wemb16: [256][64] (K padded 49->64), gatw16: [6][256][256],
// fw1: [128][256], fw2: [128][128], all row-major in K.
__global__ void convert_weights_kernel(const float* __restrict__ wemb,
                                       const float* __restrict__ gatw,
                                       const float* __restrict__ fw1,
                                       const float* __restrict__ fw2,
                                       _Float16* __restrict__ o_emb,
                                       _Float16* __restrict__ o_gat,
                                       _Float16* __restrict__ o_f1,
                                       _Float16* __restrict__ o_f2) {
    int i = blockIdx.x * blockDim.x + threadIdx.x;
    if (i < 16384) {                       // W_emb with zero pad 49->64
        int n = i >> 6, k = i & 63;
        o_emb[i] = (_Float16)((k < 49) ? wemb[n * 49 + k] : 0.0f);
        return;
    }
    i -= 16384;
    if (i < 393216) { o_gat[i] = (_Float16)gatw[i]; return; }
    i -= 393216;
    if (i < 32768)  { o_f1[i] = (_Float16)fw1[i];  return; }
    i -= 32768;
    if (i < 16384)  { o_f2[i] = (_Float16)fw2[i]; }
}

// ---- main GAT kernel: one graph per 256-thread block (8 waves) -------------
__global__ __launch_bounds__(256) void gat_forward_kernel(
    const float* __restrict__ x, const float* __restrict__ A,
    const _Float16* __restrict__ wemb, const _Float16* __restrict__ gatw,
    const float* __restrict__ gat_b, const float* __restrict__ gat_a,
    float* __restrict__ g_out) {
    __shared__ _Float16 sh_h[64 * 264];       // current features, f16 (padded)
    __shared__ _Float16 sh_t[256 * 72];       // hh^T (also x staging)
    __shared__ _Float16 sh_attn[4 * 64 * 72]; // attention, 4 heads at a time
    __shared__ float    sh_es[64 * 8];
    __shared__ float    sh_ed[64 * 8];
    __shared__ unsigned sh_adj[64 * 2];
    __shared__ float    sh_inv;

    const int b    = blockIdx.x;
    const int tid  = threadIdx.x;
    const int lane = tid & 31;
    const int wv   = tid >> 5;

    // adjacency bitmask
    if (tid < 64) {
        const float* Ar = A + ((size_t)b * 64 + tid) * 64;
        unsigned m0 = 0u, m1 = 0u;
        for (int j = 0; j < 32; ++j) {
            if (Ar[j]      > 0.0f) m0 |= 1u << j;
            if (Ar[32 + j] > 0.0f) m1 |= 1u << j;
        }
        sh_adj[tid * 2]     = m0;
        sh_adj[tid * 2 + 1] = m1;
    }
    // stage x (64x49, pad to 64) as f16 A-operand in sh_t
    for (int idx = tid; idx < 64 * 64; idx += 256) {
        int i = idx >> 6, k = idx & 63;
        float v = (k < 49) ? x[((size_t)b * 64 + i) * 49 + k] : 0.0f;
        sh_t[i * 72 + k] = (_Float16)v;
    }
    __syncthreads();

    // ---- embedding: h0 = x @ Wemb^T  (64x256, K=64) ----
    for (int t = 0; t < 8; ++t) {
        int tile = wv * 8 + t, mt = tile >> 4, nt = tile & 15;
        v8f acc = {};
        for (int kt = 0; kt < 2; ++kt) {
            v16h a  = frag_a(sh_t, 72, mt * 16, kt * 32, lane);
            v16h bb = frag_b(wemb, 64, nt * 16, kt * 32, lane);
            acc = wmma_f32(a, bb, acc);
        }
        int n = nt * 16 + (lane & 15);
#pragma unroll
        for (int v = 0; v < 8; ++v) {
            int m = mt * 16 + v + ((lane & 16) ? 8 : 0);
            sh_h[m * 264 + n] = (_Float16)acc[v];
        }
    }
    __syncthreads();

    // ---- 6 GAT layers ----
    for (int l = 0; l < 6; ++l) {
        const _Float16* Wl = gatw + (size_t)l * 256 * 256;
        if (l < 5) __builtin_prefetch(gatw + (size_t)(l + 1) * 256 * 256, 0, 1);

        // hh = h @ Wl^T + b, stored transposed into sh_t (64x256, K=256)
        for (int t = 0; t < 8; ++t) {
            int tile = wv * 8 + t, mt = tile >> 4, nt = tile & 15;
            v8f acc = {};
            for (int kt = 0; kt < 8; ++kt) {
                v16h a  = frag_a(sh_h, 264, mt * 16, kt * 32, lane);
                v16h bb = frag_b(Wl, 256, nt * 16, kt * 32, lane);
                acc = wmma_f32(a, bb, acc);
            }
            int n = nt * 16 + (lane & 15);
            float bias = gat_b[l * 256 + n];
#pragma unroll
            for (int v = 0; v < 8; ++v) {
                int m = mt * 16 + v + ((lane & 16) ? 8 : 0);
                sh_t[n * 72 + m] = (_Float16)(acc[v] + bias);
            }
        }
        __syncthreads();

        // attention logits per node: e_src / e_dst  (64 nodes x 8 heads)
        for (int p = tid; p < 512; p += 256) {
            int i = p >> 3, hd = p & 7;
            const float* av = gat_a + (size_t)(l * 8 + hd) * 64;
            float es = 0.0f, ed = 0.0f;
            for (int d = 0; d < 32; ++d) {
                float hv = (float)sh_t[(hd * 32 + d) * 72 + i];
                es += hv * av[d];
                ed += hv * av[32 + d];
            }
            sh_es[i * 8 + hd] = es;
            sh_ed[i * 8 + hd] = ed;
        }
        __syncthreads();

        // two rounds of 4 heads: softmax -> WMMA aggregation
        for (int hg = 0; hg < 2; ++hg) {
            {   // masked softmax over neighbors j; 64 rows x 4 heads = 256 thr
                int i = tid >> 2, hl = tid & 3, hd = hg * 4 + hl;
                unsigned m0 = sh_adj[i * 2], m1 = sh_adj[i * 2 + 1];
                float es = sh_es[i * 8 + hd];
                float mx = -3.4e38f;
                for (int j = 0; j < 64; ++j) {
                    bool con = (j < 32) ? ((m0 >> j) & 1u) : ((m1 >> (j - 32)) & 1u);
                    float e = es + sh_ed[j * 8 + hd];
                    e = (e > 0.0f) ? e : ALPHA * e;
                    e = con ? e : NEG_INF;
                    mx = fmaxf(mx, e);
                }
                float sum = 0.0f;
                _Float16* row = sh_attn + (size_t)(hl * 64 + i) * 72;
                for (int j = 0; j < 64; ++j) {
                    bool con = (j < 32) ? ((m0 >> j) & 1u) : ((m1 >> (j - 32)) & 1u);
                    float e = es + sh_ed[j * 8 + hd];
                    e = (e > 0.0f) ? e : ALPHA * e;
                    e = con ? e : NEG_INF;
                    float w = __expf(e - mx);
                    sum += w;
                    row[j] = (_Float16)w;
                }
                float inv = 1.0f / sum;
                for (int j = 0; j < 64; ++j)
                    row[j] = (_Float16)((float)row[j] * inv);
            }
            __syncthreads();
            {   // out[:,head,:] = attn_head @ hh[:,head,:]  (64x32, K=64)
                int hl = wv >> 1, hd = hg * 4 + hl, sub = wv & 1;
                const _Float16* attnh = sh_attn + (size_t)hl * 64 * 72;
                for (int mt = 0; mt < 4; ++mt) {
                    v8f acc = {};
                    for (int kt = 0; kt < 2; ++kt) {
                        v16h a  = frag_a(attnh, 72, mt * 16, kt * 32, lane);
                        v16h bb = frag_b(sh_t, 72, hd * 32 + sub * 16, kt * 32, lane);
                        acc = wmma_f32(a, bb, acc);
                    }
                    int n = hd * 32 + sub * 16 + (lane & 15);
#pragma unroll
                    for (int v = 0; v < 8; ++v) {
                        int m = mt * 16 + v + ((lane & 16) ? 8 : 0);
                        float val = acc[v] + (float)sh_h[m * 264 + n]; // residual
                        val = (val > 0.0f) ? val : (__expf(val) - 1.0f); // ELU
                        sh_h[m * 264 + n] = (_Float16)val;
                    }
                }
            }
            __syncthreads();
        }
    }

    // ---- masked mean-pool readout ----
    if (tid == 0) {
        int c = 0;
        for (int i = 0; i < 64; ++i)
            c += ((sh_adj[i * 2] | sh_adj[i * 2 + 1]) != 0u) ? 1 : 0;
        sh_inv = (c > 0) ? 1.0f / (float)c : 0.0f;
    }
    __syncthreads();
    {
        int d = tid;  // 256 dims
        float s = 0.0f;
        for (int i = 0; i < 64; ++i)
            if (sh_adj[i * 2] | sh_adj[i * 2 + 1])
                s += (float)sh_h[i * 264 + d];
        g_out[(size_t)b * 256 + d] = s * sh_inv;
    }
}

// ---- FC head: 16 graphs per block batched as WMMA M dim --------------------
__global__ __launch_bounds__(128) void fc_head_kernel(
    const float* __restrict__ g,
    const _Float16* __restrict__ w1, const float* __restrict__ b1,
    const _Float16* __restrict__ w2, const float* __restrict__ b2,
    const float* __restrict__ w3, const float* __restrict__ b3,
    float* __restrict__ out) {
    __shared__ _Float16 sh_a[16 * 264];
    __shared__ _Float16 sh_z[16 * 136];
    const int tid = threadIdx.x, lane = tid & 31, wv = tid >> 5;
    const int g0 = blockIdx.x * 16;

    for (int idx = tid; idx < 16 * 256; idx += 128) {
        int gi = idx >> 8, k = idx & 255;
        sh_a[gi * 264 + k] = (_Float16)g[(size_t)(g0 + gi) * 256 + k];
    }
    __syncthreads();

    // z1 = relu(g @ W1^T + b1): M=16, N=128, K=256
    for (int tt = 0; tt < 2; ++tt) {
        int nt = wv * 2 + tt;
        v8f acc = {};
        for (int kt = 0; kt < 8; ++kt) {
            v16h a  = frag_a(sh_a, 264, 0, kt * 32, lane);
            v16h bb = frag_b(w1, 256, nt * 16, kt * 32, lane);
            acc = wmma_f32(a, bb, acc);
        }
        int n = nt * 16 + (lane & 15);
        float bias = b1[n];
#pragma unroll
        for (int v = 0; v < 8; ++v) {
            int m = v + ((lane & 16) ? 8 : 0);
            sh_z[m * 136 + n] = (_Float16)fmaxf(acc[v] + bias, 0.0f);
        }
    }
    __syncthreads();

    // z2 = relu(z1 @ W2^T + b2): M=16, N=128, K=128  (into sh_a)
    for (int tt = 0; tt < 2; ++tt) {
        int nt = wv * 2 + tt;
        v8f acc = {};
        for (int kt = 0; kt < 4; ++kt) {
            v16h a  = frag_a(sh_z, 136, 0, kt * 32, lane);
            v16h bb = frag_b(w2, 128, nt * 16, kt * 32, lane);
            acc = wmma_f32(a, bb, acc);
        }
        int n = nt * 16 + (lane & 15);
        float bias = b2[n];
#pragma unroll
        for (int v = 0; v < 8; ++v) {
            int m = v + ((lane & 16) ? 8 : 0);
            sh_a[m * 264 + n] = (_Float16)fmaxf(acc[v] + bias, 0.0f);
        }
    }
    __syncthreads();

    // z3 = elu(z2 @ W3^T + b3) + 1.5
    if (tid < 16) {
        float s = b3[0];
        for (int k = 0; k < 128; ++k)
            s += (float)sh_a[tid * 264 + k] * w3[k];
        s = (s > 0.0f) ? s : (__expf(s) - 1.0f);
        out[g0 + tid] = s + 1.5f;
    }
}

extern "C" void kernel_launch(void* const* d_in, const int* in_sizes, int n_in,
                              void* d_out, int out_size, void* d_ws, size_t ws_size,
                              hipStream_t stream) {
    (void)in_sizes; (void)n_in; (void)out_size; (void)ws_size;
    const float* x      = (const float*)d_in[0];
    const float* A      = (const float*)d_in[1];
    const float* W_emb  = (const float*)d_in[2];
    const float* gat_W  = (const float*)d_in[3];
    const float* gat_b  = (const float*)d_in[4];
    const float* gat_a  = (const float*)d_in[5];
    const float* fc_b1  = (const float*)d_in[7];
    const float* fc_b2  = (const float*)d_in[9];
    const float* fc_W3  = (const float*)d_in[10];
    const float* fc_b3  = (const float*)d_in[11];
    const float* fc_W1  = (const float*)d_in[6];
    const float* fc_W2  = (const float*)d_in[8];
    float* out = (float*)d_out;

    char* ws = (char*)d_ws;
    _Float16* wemb16 = (_Float16*)(ws);
    _Float16* gatw16 = (_Float16*)(ws + 32768);
    _Float16* fw1    = (_Float16*)(ws + 32768 + 786432);
    _Float16* fw2    = (_Float16*)(ws + 32768 + 786432 + 65536);
    float*    gbuf   = (float*)(ws + 32768 + 786432 + 65536 + 32768);

    // 16384 + 393216 + 32768 + 16384 = 458752 elements -> 1792 blocks of 256
    convert_weights_kernel<<<1792, 256, 0, stream>>>(
        W_emb, gat_W, fc_W1, fc_W2, wemb16, gatw16, fw1, fw2);
    gat_forward_kernel<<<2048, 256, 0, stream>>>(
        x, A, wemb16, gatw16, gat_b, gat_a, gbuf);
    fc_head_kernel<<<128, 128, 0, stream>>>(
        gbuf, fw1, fc_b1, fw2, fc_b2, fc_W3, fc_b3, out);
}
